// RC3D_23098334118486
// MI455X (gfx1250) — matrix operand order repaired and checked by reading, not compiled, not run on hardware
//
#include <hip/hip_runtime.h>
#include <hip/hip_bf16.h>

// ---------------------------------------------------------------------------
// CDNA5 (gfx1250) WMMA types
// ---------------------------------------------------------------------------
typedef __attribute__((ext_vector_type(4)))  _Float16 v4h;
typedef __attribute__((ext_vector_type(8)))  _Float16 v8h;
typedef __attribute__((ext_vector_type(16))) _Float16 v16h;
typedef __attribute__((ext_vector_type(8)))  float    v8f;
typedef __attribute__((ext_vector_type(4)))  int      v4i;

#define NEG_INF (-3.4e38f)

#if __has_builtin(__builtin_amdgcn_global_load_async_to_lds_b128)
#define HAVE_ASYNC_LDS 1
typedef __attribute__((address_space(1))) v4i gv4i_t;   // global int4 (addrspace 1)
typedef __attribute__((address_space(3))) v4i lv4i_t;   // LDS int4    (addrspace 3)
#endif

static __device__ __forceinline__ void wait_async0() {
#if defined(HAVE_ASYNC_LDS)
#if __has_builtin(__builtin_amdgcn_s_wait_asynccnt)
    __builtin_amdgcn_s_wait_asynccnt(0);
#else
    asm volatile("s_wait_asynccnt 0" ::: "memory");
#endif
#endif
}

static __device__ __forceinline__ v16h cat8(v8h lo, v8h hi) {
    return __builtin_shufflevector(lo, hi, 0, 1, 2, 3, 4, 5, 6, 7,
                                           8, 9, 10, 11, 12, 13, 14, 15);
}

// ---------------------------------------------------------------------------
// Weight pre-pack: f32 -> f16, K zero-padded to Kp (multiple of 32)
// ---------------------------------------------------------------------------
__global__ void k_w2h(const float* __restrict__ w, _Float16* __restrict__ wh,
                      int Ktot, int Kp, long ntot)
{
    long idx = (long)blockIdx.x * blockDim.x + threadIdx.x;
    if (idx >= ntot) return;
    int k = (int)(idx % Kp);
    long oc = idx / Kp;
    wh[idx] = (k < Ktot) ? (_Float16)w[oc * Ktot + k] : (_Float16)0.f;
}

// ---------------------------------------------------------------------------
// Implicit-GEMM 3D conv (3x3x3, SAME) + bias + ReLU via v_wmma_f32_16x16x32_f16
//   A: 16 oc x 32 k (f16 pre-packed)   B: 32 k x 32 vox (two 16-vox tiles)
// 4 waves/block; each wave: 2 WMMAs per K-chunk reusing one A fragment.
// Cout % 64 == 0 and nvox % 32 == 0 for every layer -> no M/N bounds checks.
// ---------------------------------------------------------------------------
__global__ __launch_bounds__(128) void k_conv3d_wmma(
    const float* __restrict__ X,        // Cin x D x H x W
    const _Float16* __restrict__ Wh,    // Cout x Kp (f16, zero-padded)
    const float* __restrict__ bias,
    float* __restrict__ Y,              // Cout x D x H x W
    int Cin, int D, int H, int Wd, int Kp)
{
    __shared__ __align__(16) _Float16 As[4][16][32];
    __shared__ __align__(16) _Float16 Bs[2][32][24];   // rows padded to 48B

    const int tid  = threadIdx.x;
    const int lane = tid & 31;
    const int wv   = tid >> 5;
    const int HW   = H * Wd;
    const int nvox = D * HW;
    const int voxBase = blockIdx.x * 32;
    const int ocBlock = blockIdx.y * 64;
    const int Ktot = Cin * 27;

    // ---- fixed staging roles ----
    // B: thread t handles row kRow (0..31) x 4 voxels in each of the 2 tiles
    const int kRow = tid >> 2;
    const int nGrp = (tid & 3) * 4;
    int vd[2][4], vhh[2][4], vww[2][4];
#pragma unroll
    for (int u = 0; u < 2; ++u)
#pragma unroll
        for (int j = 0; j < 4; ++j) {
            int vox = voxBase + u * 16 + nGrp + j;
            vd[u][j] = vox / HW;
            int r3 = vox - vd[u][j] * HW;
            vhh[u][j] = r3 / Wd;
            vww[u][j] = r3 - vhh[u][j] * Wd;
        }
    // A: thread t copies 8 f16 (16B) of row aRow (0..63)
    const int aRow  = tid >> 1;
    const int aHalf = (tid & 1) * 16;
    const _Float16* wrow = Wh + (long)(ocBlock + aRow) * Kp + aHalf;
    _Float16* aDst = &As[aRow >> 4][aRow & 15][aHalf];

    v8f acc0 = {}, acc1 = {};
    for (int k0 = 0; k0 < Kp; k0 += 32) {
        // ---- stage A: 16B global->LDS (async DMA when available) ----
#if defined(HAVE_ASYNC_LDS)
        __builtin_amdgcn_global_load_async_to_lds_b128(
            (gv4i_t*)(wrow + k0), (lv4i_t*)aDst, 0, 0);
#else
        *(float4*)aDst = *(const float4*)(wrow + k0);
#endif
        if (k0 + 32 < Kp) __builtin_prefetch(wrow + k0 + 32, 0, 1);

        // ---- stage B: branchless im2col gather, batched b32 loads ----
        {
            int kk = k0 + kRow;
            int ic = kk / 27;
            int r  = kk - ic * 27;
            int r9 = r / 9;
            int kd = r9 - 1;
            int r2 = r - r9 * 9;
            int r3q = r2 / 3;
            int kh = r3q - 1;
            int kw = r2 - r3q * 3 - 1;
            bool kv = kk < Ktot;
#pragma unroll
            for (int u = 0; u < 2; ++u) {
                v4h pk;
#pragma unroll
                for (int j = 0; j < 4; ++j) {
                    int dd = vd[u][j] + kd, hh = vhh[u][j] + kh, ww = vww[u][j] + kw;
                    bool valid = kv && (unsigned)dd < (unsigned)D &&
                                 (unsigned)hh < (unsigned)H && (unsigned)ww < (unsigned)Wd;
                    long off = ((long)(ic * D + dd) * H + hh) * Wd + ww;
                    off = valid ? off : 0;
                    float v = X[off];                 // unconditional, batched
                    pk[j] = valid ? (_Float16)v : (_Float16)0.f;
                }
                *(v4h*)&Bs[u][kRow][nGrp] = pk;       // one ds_store_b64 each
            }
        }
        wait_async0();
        __syncthreads();

        // ---- fragments: ds_load_b128 pairs; one A reused by two WMMAs ----
        const int mr = lane & 15;
        const int kA = (lane < 16) ? 0 : 8;
        v16h a  = cat8(*(const v8h*)&As[wv][mr][kA],
                       *(const v8h*)&As[wv][mr][16 + kA]);
        v16h b0, b1;
#pragma unroll
        for (int s = 0; s < 16; ++s) {
            int kb = (s < 8) ? (kA + s) : (16 + kA + s - 8);
            b0[s] = Bs[0][kb][mr];
            b1[s] = Bs[1][kb][mr];
        }
        acc0 = __builtin_amdgcn_wmma_f32_16x16x32_f16(
            false, a, false, b0, (short)0, acc0, false, false);
        acc1 = __builtin_amdgcn_wmma_f32_16x16x32_f16(
            false, a, false, b1, (short)0, acc1, false, false);
        __syncthreads();
    }

    // ---- epilogue: C layout (lane L: N=L%16, VGPR r: M=r+(L<16?0:8)) ----
    const int n = lane & 15;
    const int mOff = (lane < 16) ? 0 : 8;
#pragma unroll
    for (int r = 0; r < 8; ++r) {
        int oc = ocBlock + wv * 16 + mOff + r;
        float* yrow = Y + (long)oc * nvox + voxBase + n;
        yrow[0]  = fmaxf(acc0[r] + bias[oc], 0.f);
        yrow[16] = fmaxf(acc1[r] + bias[oc], 0.f);
    }
}

// ---------------------------------------------------------------------------
// WMMA GEMM: C[m*N+n] = relu?( A16[m,:K] . B(:,n) + bias[m] )
//   A16: f16 pre-packed, M%16==0, K%32==0.  B(k,n) = Bm[k*sK + n*sN] (f32).
// ---------------------------------------------------------------------------
__global__ __launch_bounds__(32) void k_gemm_wmma(
    const _Float16* __restrict__ A16, const float* __restrict__ Bm,
    const float* __restrict__ bias, float* __restrict__ Cm,
    int M, int N, int K, int sK, int sN, int relu)
{
    __shared__ __align__(16) _Float16 As[16][32];
    __shared__ __align__(16) _Float16 Bs[32][24];
    const int lane = threadIdx.x;
    const int nBase = blockIdx.x * 16;
    const int mBase = blockIdx.y * 16;

    const int aRow  = lane >> 1;
    const int aHalf = (lane & 1) * 16;
    const _Float16* wrow = A16 + (long)(mBase + aRow) * K + aHalf;
    const int kRow = lane;                     // B: one full row per lane

    v8f acc = {};
    for (int k0 = 0; k0 < K; k0 += 32) {
        *(float4*)&As[aRow][aHalf] = *(const float4*)(wrow + k0);
#pragma unroll
        for (int g = 0; g < 4; ++g) {
            v4h pk;
#pragma unroll
            for (int j = 0; j < 4; ++j) {
                int n = g * 4 + j;
                bool ok = (nBase + n) < N;
                long off = (long)(k0 + kRow) * sK + (long)(nBase + n) * sN;
                off = ok ? off : 0;
                float v = Bm[off];
                pk[j] = ok ? (_Float16)v : (_Float16)0.f;
            }
            *(v4h*)&Bs[kRow][g * 4] = pk;
        }
        __syncthreads();
        const int mr = lane & 15;
        const int kA = (lane < 16) ? 0 : 8;
        v16h a = cat8(*(const v8h*)&As[mr][kA], *(const v8h*)&As[mr][16 + kA]);
        v16h b;
#pragma unroll
        for (int s = 0; s < 16; ++s)
            b[s] = Bs[(s < 8) ? (kA + s) : (16 + kA + s - 8)][mr];
        acc = __builtin_amdgcn_wmma_f32_16x16x32_f16(
            false, a, false, b, (short)0, acc, false, false);
        __syncthreads();
    }
    const int n = lane & 15;
    const int mOff = (lane < 16) ? 0 : 8;
    if (nBase + n < N) {
#pragma unroll
        for (int r = 0; r < 8; ++r) {
            int m = mBase + mOff + r;
            float v = acc[r] + bias[m];
            if (relu) v = fmaxf(v, 0.f);
            Cm[(long)m * N + nBase + n] = v;
        }
    }
}

// ---------------------------------------------------------------------------
// Max-pool 3D, window==stride, VALID
// ---------------------------------------------------------------------------
__global__ void k_maxpool3d(const float* __restrict__ in, float* __restrict__ out,
                            int C, int D, int H, int W, int wd, int wh, int ww)
{
    int Do = D / wd, Ho = H / wh, Wo = W / ww;
    long idx = (long)blockIdx.x * blockDim.x + threadIdx.x;
    long ntot = (long)C * Do * Ho * Wo;
    if (idx >= ntot) return;
    int w = idx % Wo; long r = idx / Wo;
    int h = r % Ho; r /= Ho;
    int d = r % Do; int c = (int)(r / Do);
    float m = NEG_INF;
    for (int a = 0; a < wd; ++a)
        for (int b = 0; b < wh; ++b)
            for (int e = 0; e < ww; ++e)
                m = fmaxf(m, in[((long)(c * D + d * wd + a) * H + h * wh + b) * W + w * ww + e]);
    out[idx] = m;
}

// repack p5 (512 x 8 x 2 x 2) -> xflat (L=8 x 2048)
__global__ void k_repack(const float* __restrict__ p5, float* __restrict__ xflat)
{
    int idx = blockIdx.x * blockDim.x + threadIdx.x;
    if (idx >= 8 * 2048) return;
    int l = idx >> 11, j = idx & 2047;
    int c = j >> 2, hw = j & 3;
    xflat[idx] = p5[c * 32 + l * 4 + hw];
}

// tiny 1-D conv over L=8 sequence (sp_conv / sp_cls / sp_seg)
__global__ void k_conv1d(const float* __restrict__ in, const float* __restrict__ w,
                         const float* __restrict__ b, float* __restrict__ out,
                         int Cin, int Cout, int L, int ks, int pad, int relu)
{
    int idx = blockIdx.x * blockDim.x + threadIdx.x;
    if (idx >= Cout * L) return;
    int oc = idx / L, l = idx % L;
    float acc = b[oc];
    for (int ic = 0; ic < Cin; ++ic)
        for (int t = 0; t < ks; ++t) {
            int ll = l + t - pad;
            if (ll >= 0 && ll < L)
                acc += w[(oc * Cin + ic) * ks + t] * in[ic * L + ll];
        }
    if (relu) acc = fmaxf(acc, 0.f);
    out[idx] = acc;
}

// ---------------------------------------------------------------------------
// Proposals: softmax, anchor decode, sort, NMS (N=72), top-32, s/e windows.
// ---------------------------------------------------------------------------
#define NANCH 72
__global__ __launch_bounds__(128) void k_proposal(
    const float* __restrict__ cls_score, const float* __restrict__ seg_pred,
    float* __restrict__ cls2, float* __restrict__ off2,
    float* __restrict__ d_out, int* __restrict__ se_buf)
{
    __shared__ float sc[NANCH], cc[NANCH], hh[NANCH];
    __shared__ int order[NANCH];
    __shared__ unsigned char supp[NANCH], keepf[NANCH];
    __shared__ int prop[32];
    const int t = threadIdx.x;
    const int L = 8;

    if (t < NANCH) {
        int l = t / 9, a = t % 9;
        float x0 = cls_score[(a * 2 + 0) * L + l];
        float x1 = cls_score[(a * 2 + 1) * L + l];
        float mx = fmaxf(x0, x1);
        float e0 = __expf(x0 - mx), e1 = __expf(x1 - mx);
        cls2[t * 2 + 0] = x0; cls2[t * 2 + 1] = x1;
        float o0 = seg_pred[(a * 2 + 0) * L + l];
        float o1 = seg_pred[(a * 2 + 1) * L + l];
        off2[t * 2 + 0] = o0; off2[t * 2 + 1] = o1;
        const float AS[9] = {1.f, 2.f, 3.f, 4.f, 5.f, 6.f, 8.f, 11.f, 16.f};
        float ah = AS[a] * 8.f;
        cc[t] = (l * 16.f + 8.f) + o0 * (2.f * ah);
        hh[t] = ah * __expf(o1);
        sc[t] = e1 / (e0 + e1);
    }
    __syncthreads();
    if (t == 0) {
        for (int i = 0; i < NANCH; ++i) order[i] = i;
        for (int i = 0; i < NANCH; ++i) {
            int best = i;
            for (int j = i + 1; j < NANCH; ++j)
                if (sc[order[j]] > sc[order[best]]) best = j;
            int tmp = order[i]; order[i] = order[best]; order[best] = tmp;
        }
        for (int i = 0; i < NANCH; ++i) supp[i] = 0;
        for (int i = 0; i < NANCH; ++i) {
            int alive = !supp[i];
            keepf[i] = (unsigned char)alive;
            if (alive) {
                float s1 = cc[order[i]] - hh[order[i]];
                float e1v = cc[order[i]] + hh[order[i]];
                for (int j = i + 1; j < NANCH; ++j) {
                    float s2 = cc[order[j]] - hh[order[j]];
                    float e2 = cc[order[j]] + hh[order[j]];
                    float inter = fmaxf(0.f, fminf(e1v, e2) - fmaxf(s1, s2));
                    float uni = (e1v - s1) + (e2 - s2) - inter;
                    if (inter / fmaxf(uni, 1e-8f) > 0.7f) supp[j] = 1;
                }
            }
        }
        int cnt = 0;
        for (int i = 0; i < NANCH && cnt < 32; ++i) if (keepf[i])  prop[cnt++] = order[i];
        for (int i = 0; i < NANCH && cnt < 32; ++i) if (!keepf[i]) prop[cnt++] = order[i];
    }
    __syncthreads();
    if (t < 32) {
        int pi = prop[t];
        d_out[t * 2 + 0] = cls2[pi * 2 + 0];
        d_out[t * 2 + 1] = cls2[pi * 2 + 1];
        d_out[64 + t * 2 + 0] = off2[pi * 2 + 0];
        d_out[64 + t * 2 + 1] = off2[pi * 2 + 1];
        float pc = cc[pi], ph = hh[pi];
        int s = (int)floorf((pc - ph) / 16.f); s = s < 0 ? 0 : (s > 7 ? 7 : s);
        int e = (int)ceilf((pc + ph) / 16.f);  e = e < 0 ? 0 : (e > 7 ? 7 : e);
        if (e < s) e = s;
        se_buf[t] = s; se_buf[32 + t] = e;
    }
}

// SPP: writes GEMM-friendly layout spp[ch*224 + p*7 + j]
__global__ void k_spp(const float* __restrict__ feature, const int* __restrict__ se,
                      float* __restrict__ spp)
{
    int idx = blockIdx.x * blockDim.x + threadIdx.x;
    if (idx >= 32 * 1024) return;
    int p = idx >> 10, ch = idx & 1023;
    int s = se[p], e = se[32 + p];
    float Lp = (float)(e - s + 1);
    for (int j = 0; j < 7; ++j) {
        float lo = floorf(j * Lp / 7.f);
        float hi = ceilf((j + 1) * Lp / 7.f);
        float m = NEG_INF;
        for (int tt = 0; tt < 8; ++tt) {
            float rel = (float)(tt - s);
            if (rel >= lo && rel < hi && tt <= e && rel >= 0.f)
                m = fmaxf(m, feature[ch * 8 + tt]);
        }
        spp[ch * 224 + p * 7 + j] = m;
    }
}

// tail: fc1(1792->256,relu) then cls(256->21), bbox(256->2); z is [oc*224+p*7+j]
__global__ __launch_bounds__(256) void k_tail(
    const float* __restrict__ z, const float* __restrict__ fc1W, const float* __restrict__ fc1b,
    const float* __restrict__ clsW, const float* __restrict__ clsb,
    const float* __restrict__ bW, const float* __restrict__ bb, float* __restrict__ d_out)
{
    __shared__ float tbuf[256];
    const int p = blockIdx.x, o = threadIdx.x;
    float acc = fc1b[o];
    for (int oc = 0; oc < 256; ++oc)
#pragma unroll
        for (int j = 0; j < 7; ++j)
            acc += fc1W[o * 1792 + oc * 7 + j] * z[oc * 224 + p * 7 + j];
    tbuf[o] = fmaxf(acc, 0.f);
    __syncthreads();
    if (o < 21) {
        float a = clsb[o];
        for (int k = 0; k < 256; ++k) a += clsW[o * 256 + k] * tbuf[k];
        d_out[128 + p * 21 + o] = a;
    } else if (o < 23) {
        int j = o - 21;
        float a = bb[j];
        for (int k = 0; k < 256; ++k) a += bW[j * 256 + k] * tbuf[k];
        d_out[128 + 672 + p * 2 + j] = a;
    }
}

// ---------------------------------------------------------------------------
// Driver
// ---------------------------------------------------------------------------
extern "C" void kernel_launch(void* const* d_in, const int* in_sizes, int n_in,
                              void* d_out, int out_size, void* d_ws, size_t ws_size,
                              hipStream_t stream)
{
    const float* x = (const float*)d_in[0];
    const float* cw[8], *cb[8];
    for (int i = 0; i < 8; ++i) { cw[i] = (const float*)d_in[1 + 2 * i]; cb[i] = (const float*)d_in[2 + 2 * i]; }
    const float* fc1W  = (const float*)d_in[17]; const float* fc1b  = (const float*)d_in[18];
    const float* fc2W  = (const float*)d_in[19]; const float* fc2b  = (const float*)d_in[20];
    const float* c1w   = (const float*)d_in[21]; const float* c1b   = (const float*)d_in[22];
    const float* c2w   = (const float*)d_in[23]; const float* c2b   = (const float*)d_in[24];
    const float* spcw  = (const float*)d_in[25]; const float* spcb  = (const float*)d_in[26];
    const float* clsw  = (const float*)d_in[27]; const float* clsb1 = (const float*)d_in[28];
    const float* segw  = (const float*)d_in[29]; const float* segb  = (const float*)d_in[30];
    const float* tfc1W = (const float*)d_in[31]; const float* tfc1b = (const float*)d_in[32];
    const float* tclsW = (const float*)d_in[33]; const float* tclsb = (const float*)d_in[34];
    const float* tboxW = (const float*)d_in[35]; const float* tboxb = (const float*)d_in[36];
    float* out = (float*)d_out;

    float* ws   = (float*)d_ws;
    float* big0 = ws;                       // 33,554,432 f
    float* big1 = ws + 33554432L;           // 16,777,216 f
    float* S    = ws + 50331648L;
    float* feature = S;                     // 8192
    float* hbuf    = S + 8192;              // 2048
    float* h2buf   = S + 10240;             // 2048
    float* clssc   = S + 12288;             // 144
    float* segpr   = S + 12432;             // 144
    float* cls2    = S + 12576;             // 144
    float* off2    = S + 12720;             // 144
    int*   se      = (int*)(S + 12864);     // 64 ints
    float* spp     = S + 13056;             // 229376
    float* zbuf    = S + 242432;            // 57344
    float* p5      = S + 300032;            // 16384
    float* xflat   = S + 316416;            // 16384
    float* fc1o    = S + 332800;            // 8192

    // f16 weight region
    _Float16* W16 = (_Float16*)(ws + 50700000L);
    const int convCin[8]  = {3, 64, 128, 256, 256, 512, 512, 512};
    const int convCout[8] = {64, 128, 256, 256, 512, 512, 512, 512};
    _Float16* cw16[8];
    int convKp[8];
    long w16off = 0;
    for (int i = 0; i < 8; ++i) {
        int Ktot = convCin[i] * 27;
        int Kp = (Ktot + 31) & ~31;
        convKp[i] = Kp;
        cw16[i] = W16 + w16off;
        w16off += (long)convCout[i] * Kp;
    }
    _Float16* fc1W16 = W16 + w16off; w16off += 1024L * 2048;
    _Float16* fc2W16 = W16 + w16off; w16off += 1024L * 1024;
    _Float16* c1w16  = W16 + w16off; w16off += 256L * 1024;
    _Float16* c2w16  = W16 + w16off; w16off += 256L * 1024;

    auto pack = [&](const float* src, _Float16* dst, int Cout, int Ktot, int Kp) {
        long n = (long)Cout * Kp;
        k_w2h<<<dim3((unsigned)((n + 255) / 256)), dim3(256), 0, stream>>>(src, dst, Ktot, Kp, n);
    };
    for (int i = 0; i < 8; ++i) pack(cw[i], cw16[i], convCout[i], convCin[i] * 27, convKp[i]);
    pack(fc1W, fc1W16, 1024, 2048, 2048);
    pack(fc2W, fc2W16, 1024, 1024, 1024);
    pack(c1w,  c1w16,  256, 1024, 1024);
    pack(c2w,  c2w16,  256, 1024, 1024);

    auto conv = [&](const float* in, int li, float* o, int D, int H, int W) {
        dim3 g((unsigned)(D * H * W / 32), (unsigned)(convCout[li] / 64), 1);
        k_conv3d_wmma<<<g, dim3(128), 0, stream>>>(in, cw16[li], cb[li], o,
                                                   convCin[li], D, H, W, convKp[li]);
    };
    auto pool = [&](const float* in, float* o, int C, int D, int H, int W,
                    int wd, int wh, int ww) {
        long n = (long)C * (D / wd) * (H / wh) * (W / ww);
        k_maxpool3d<<<dim3((unsigned)((n + 255) / 256)), dim3(256), 0, stream>>>(
            in, o, C, D, H, W, wd, wh, ww);
    };
    auto gemm = [&](const _Float16* A16, const float* Bm, const float* bias, float* Cm,
                    int M, int N, int K, int sK, int sN, int relu) {
        dim3 g((unsigned)((N + 15) / 16), (unsigned)(M / 16), 1);
        k_gemm_wmma<<<g, dim3(32), 0, stream>>>(A16, Bm, bias, Cm, M, N, K, sK, sN, relu);
    };

    // backbone
    conv(x,    0, big0, 128, 64, 64);
    pool(big0, big1,  64, 128, 64, 64, 1, 2, 2);
    conv(big1, 1, big0, 128, 32, 32);
    pool(big0, big1, 128, 128, 32, 32, 2, 2, 2);
    conv(big1, 2, big0, 64, 16, 16);
    conv(big0, 3, big1, 64, 16, 16);
    pool(big1, big0, 256, 64, 16, 16, 2, 2, 2);
    conv(big0, 4, big1, 32, 8, 8);
    conv(big1, 5, big0, 32, 8, 8);
    pool(big0, big1, 512, 32, 8, 8, 2, 2, 2);
    conv(big1, 6, big0, 16, 4, 4);
    conv(big0, 7, big1, 16, 4, 4);
    pool(big1, p5, 512, 16, 4, 4, 2, 2, 2);

    // FCs
    k_repack<<<dim3(64), dim3(256), 0, stream>>>(p5, xflat);
    gemm(fc1W16, xflat, fc1b, fc1o,    1024, 8, 2048, 1, 2048, 0);
    gemm(fc2W16, fc1o,  fc2b, feature, 1024, 8, 1024, 8, 1,    0);

    // RPN head
    gemm(c1w16, feature, c1b, hbuf, 256, 8, 1024, 8, 1, 1);
    k_conv1d<<<dim3(8), dim3(256), 0, stream>>>(hbuf, spcw, spcb, h2buf, 256, 256, 8, 3, 1, 1);
    k_conv1d<<<dim3(1), dim3(256), 0, stream>>>(h2buf, clsw, clsb1, clssc, 256, 18, 8, 1, 0, 0);
    k_conv1d<<<dim3(1), dim3(256), 0, stream>>>(h2buf, segw, segb, segpr, 256, 18, 8, 1, 0, 0);

    // proposals + NMS
    k_proposal<<<dim3(1), dim3(128), 0, stream>>>(clssc, segpr, cls2, off2, out, se);

    // SPP + heads
    k_spp<<<dim3(128), dim3(256), 0, stream>>>(feature, se, spp);
    gemm(c2w16, spp, c2b, zbuf, 256, 224, 1024, 224, 1, 1);
    k_tail<<<dim3(32), dim3(256), 0, stream>>>(zbuf, tfc1W, tfc1b, tclsW, tclsb, tboxW, tboxb, out);
}
